// SemanticAttention_21122649162372
// MI455X (gfx1250) — compile-verified, matrix operand order checked by reading, not run on hardware
//
#include <hip/hip_runtime.h>

// ---------------- problem constants ----------------
constexpr int BB = 4;
constexpr int NN = 2048;
constexpr int DD = 768;
constexpr long long BN   = (long long)BB * NN;        // 8192
constexpr long long NSQ  = (long long)NN * NN;        // 4194304
constexpr float LN_EPS = 1e-5f;

// ---------------- vector types ----------------
typedef __attribute__((ext_vector_type(16))) _Float16 v16h;
typedef __attribute__((ext_vector_type(8)))  _Float16 v8h;
typedef __attribute__((ext_vector_type(8)))  float    v8f;

union H16 { v16h v; v8h h8[2]; _Float16 h[16]; };

// ---------------- GEMM tiling ----------------
// C[M,Nt] = epilogue( A[M,K] (row major) * Bt[Nt,K]^T ), f16 in, fp32 accum.
constexpr int TM = 128, TN = 128, TK = 32;
constexpr int LSTR = 40;            // LDS row stride in halfs (pad vs bank conflicts)
constexpr int ABUF = TM * LSTR;     // halfs per A buffer
constexpr int BBUF = TN * LSTR;     // halfs per B buffer

enum { EPI_F32 = 0, EPI_BIAS_H = 1, EPI_BIAS_HT = 2, EPI_WMIX = 3, EPI_H = 4, EPI_BIAS_F32 = 5 };

struct GemmP {
  const _Float16* A;   long long sA;  int lda;
  const _Float16* Bt;  long long sB;  int ldb;
  int K;
  float alpha;
  const float* bias;                       // [Nt]
  const float* Cin;  long long sC; int ldc;     // WMIX: tanh-dist matrix
  const float* rowmax; long long sRM;           // WMIX
  const float* lamP; float scale;               // WMIX: alpha=(1-lam)*scale
  float*     outF;  long long sF;  int ldf;
  _Float16*  outH;  long long sH;  int ldh;
  _Float16*  outHT; long long sHT; int ldht; long long rowsPerBatchT; // vT store
};

// CDNA5 async global -> LDS copy, 16 bytes per lane.  Tracked by ASYNCcnt.
__device__ __forceinline__ void async_copy16(unsigned lds_off, const void* gaddr) {
  asm volatile("global_load_async_to_lds_b128 %0, %1, off"
               :: "v"(lds_off), "v"(gaddr)
               : "memory");
}
__device__ __forceinline__ void wait_async_le4() {
  asm volatile("s_wait_asynccnt 4" ::: "memory");
}
__device__ __forceinline__ void wait_async_0() {
  asm volatile("s_wait_asynccnt 0" ::: "memory");
}

template <int MODE>
__global__ __launch_bounds__(256) void gemm_wmma(GemmP p) {
  __shared__ _Float16 As[2][ABUF];
  __shared__ _Float16 Bs[2][BBUF];

  const int tid  = threadIdx.x;
  const int lane = tid & 31;
  const int wave = tid >> 5;
  const int wm = (wave >> 2) * 64;     // wave M offset inside block tile
  const int wn = (wave & 3) * 32;      // wave N offset
  const int lh = lane >> 4;            // lane K-half select
  const int lm = lane & 15;            // lane row/col select

  const int z = blockIdx.z;
  const long long blockM = (long long)blockIdx.y * TM;
  const long long blockN = (long long)blockIdx.x * TN;

  const _Float16* Ab = p.A  + (long long)z * p.sA;
  const _Float16* Bb = p.Bt + (long long)z * p.sB;

  // per-thread staging chunks: 512 chunks of 8 halfs per 128x32 tile, 2 per thread
  int cr[2], cc[2];
  unsigned aOff[2], bOff[2];
  const unsigned ldsA0 = (unsigned)(size_t)(&As[0][0]);
  const unsigned ldsB0 = (unsigned)(size_t)(&Bs[0][0]);
#pragma unroll
  for (int i = 0; i < 2; ++i) {
    int chunk = tid + i * 256;               // 0..511
    cr[i] = chunk >> 2;                      // row 0..127
    cc[i] = (chunk & 3) * 8;                 // col 0,8,16,24
    aOff[i] = (unsigned)((cr[i] * LSTR + cc[i]) * 2);
    bOff[i] = aOff[i];
  }

  v8f zero = {};
  v8f acc[4][2];
#pragma unroll
  for (int im = 0; im < 4; ++im)
#pragma unroll
    for (int in = 0; in < 2; ++in) acc[im][in] = zero;

  const int nk = p.K / TK;

  auto issue = [&](int ks, int buf) {
    const long long k0 = (long long)ks * TK;
#pragma unroll
    for (int i = 0; i < 2; ++i) {
      async_copy16(ldsA0 + (unsigned)buf * (ABUF * 2) + aOff[i],
                   &Ab[(blockM + cr[i]) * p.lda + k0 + cc[i]]);
      async_copy16(ldsB0 + (unsigned)buf * (BBUF * 2) + bOff[i],
                   &Bb[(blockN + cr[i]) * p.ldb + k0 + cc[i]]);
    }
  };

  issue(0, 0);

  for (int ks = 0; ks < nk; ++ks) {
    const int buf = ks & 1;
    if (ks + 1 < nk) {
      issue(ks + 1, buf ^ 1);   // prefetch next tile (buffer last read 1 iter ago)
      wait_async_le4();         // first 4 asyncs (current tile) retired, in order
    } else {
      wait_async_0();
    }
    __syncthreads();            // all waves' async LDS writes for `buf` visible

    H16 af[4], bf[2];
#pragma unroll
    for (int im = 0; im < 4; ++im) {
      const _Float16* src = &As[buf][(wm + im * 16 + lm) * LSTR];
      af[im].h8[0] = *(const v8h*)&src[lh * 8];
      af[im].h8[1] = *(const v8h*)&src[16 + lh * 8];
    }
#pragma unroll
    for (int in = 0; in < 2; ++in) {
      const _Float16* src = &Bs[buf][(wn + in * 16 + lm) * LSTR];
      bf[in].h8[0] = *(const v8h*)&src[lh * 8];
      bf[in].h8[1] = *(const v8h*)&src[16 + lh * 8];
    }

#pragma unroll
    for (int im = 0; im < 4; ++im)
#pragma unroll
      for (int in = 0; in < 2; ++in)
        acc[im][in] = __builtin_amdgcn_wmma_f32_16x16x32_f16(
            false, af[im].v, false, bf[in].v, (short)0, acc[im][in], false, false);

    __syncthreads();            // all reads of `buf` done before it is re-targeted
  }

  // ---------------- epilogue ----------------
  float lam = 0.f, alpha = p.alpha;
  if constexpr (MODE == EPI_WMIX) {
    lam = *p.lamP;
    alpha = (1.0f - lam) * p.scale;
  }

#pragma unroll
  for (int im = 0; im < 4; ++im) {
#pragma unroll
    for (int in = 0; in < 2; ++in) {
      const long long n = blockN + wn + in * 16 + lm;
#pragma unroll
      for (int j = 0; j < 8; ++j) {
        const long long m = blockM + wm + im * 16 + 8 * lh + j;
        float v = acc[im][in][j] * alpha;
        if constexpr (MODE == EPI_F32) {
          p.outF[(long long)z * p.sF + m * p.ldf + n] = v;
        } else if constexpr (MODE == EPI_BIAS_H) {
          v += p.bias[n];
          p.outH[(long long)z * p.sH + m * p.ldh + n] = (_Float16)v;
        } else if constexpr (MODE == EPI_BIAS_HT) {
          v += p.bias[n];
          long long b = m / p.rowsPerBatchT;
          long long r = m % p.rowsPerBatchT;
          p.outHT[b * p.sHT + n * (long long)p.ldht + r] = (_Float16)v;
        } else if constexpr (MODE == EPI_WMIX) {
          float w  = p.Cin[(long long)z * p.sC + m * p.ldc + n];
          float rm = p.rowmax[(long long)z * p.sRM + m];
          float wn_ = w / fmaxf(rm, 1e-30f) + ((m == n) ? 1.0f : 0.0f);
          p.outF[(long long)z * p.sF + m * p.ldf + n] = v + lam * wn_;
        } else if constexpr (MODE == EPI_H) {
          p.outH[(long long)z * p.sH + m * p.ldh + n] = (_Float16)v;
        } else if constexpr (MODE == EPI_BIAS_F32) {
          v += p.bias[n];
          p.outF[(long long)z * p.sF + m * p.ldf + n] = v;
        }
      }
    }
  }
}

// ---------------- block reductions ----------------
__device__ __forceinline__ float blockSum(float v, float* sh) {
  int t = threadIdx.x;
  sh[t] = v; __syncthreads();
  for (int s = 128; s > 0; s >>= 1) { if (t < s) sh[t] += sh[t + s]; __syncthreads(); }
  float r = sh[0]; __syncthreads();
  return r;
}
__device__ __forceinline__ float blockMax(float v, float* sh) {
  int t = threadIdx.x;
  sh[t] = v; __syncthreads();
  for (int s = 128; s > 0; s >>= 1) { if (t < s) sh[t] = fmaxf(sh[t], sh[t + s]); __syncthreads(); }
  float r = sh[0]; __syncthreads();
  return r;
}

// ---------------- LN + f16 convert + row sum-of-squares ----------------
// block row in [0, 2*B*N): first half -> w2v (lnq), second half -> x (lnk, lnv)
__global__ __launch_bounds__(256) void ln_prep(
    const float* __restrict__ w2v, const float* __restrict__ x,
    const float* gq, const float* bq, const float* gk, const float* bk,
    const float* gv, const float* bv,
    _Float16* w2v_h, _Float16* x_h,
    _Float16* lnq_h, _Float16* lnk_h, _Float16* lnv_h,
    float* s2a, float* s2b) {
  __shared__ float sh[256];
  const long long row = blockIdx.x;
  const bool isX = row >= BN;
  const long long r = isX ? row - BN : row;
  const float* src = (isX ? x : w2v) + r * DD;
  const int t = threadIdx.x;

  float e[3], sum = 0.f, sq = 0.f;
#pragma unroll
  for (int i = 0; i < 3; ++i) {
    e[i] = src[t + i * 256];
    sum += e[i]; sq += e[i] * e[i];
  }
  float tsum = blockSum(sum, sh);
  float tsq  = blockSum(sq, sh);
  const float mean = tsum / (float)DD;
  const float var  = tsq / (float)DD - mean * mean;
  const float rstd = rsqrtf(var + LN_EPS);

  _Float16* raw = (isX ? x_h : w2v_h) + r * DD;
#pragma unroll
  for (int i = 0; i < 3; ++i) {
    int c = t + i * 256;
    raw[c] = (_Float16)e[i];
    float ln = (e[i] - mean) * rstd;
    if (!isX) {
      lnq_h[r * DD + c] = (_Float16)(ln * gq[c] + bq[c]);
    } else {
      lnk_h[r * DD + c] = (_Float16)(ln * gk[c] + bk[c]);
      lnv_h[r * DD + c] = (_Float16)(ln * gv[c] + bv[c]);
    }
  }
  if (t == 0) (isX ? s2b : s2a)[r] = tsq;
}

// ---------------- weight transpose + f16 convert (4 matrices) ----------------
__global__ __launch_bounds__(256) void prep_weights(
    const float* Wq, const float* Wk, const float* Wv, const float* Wp,
    _Float16* WqT, _Float16* WkT, _Float16* WvT, _Float16* WpT) {
  long long idx = (long long)blockIdx.x * 256 + threadIdx.x;
  if (idx >= 4LL * DD * DD) return;
  int w = (int)(idx / (DD * DD));
  long long rem = idx % (DD * DD);
  int i = (int)(rem / DD), j = (int)(rem % DD);
  const float* src = (w == 0) ? Wq : (w == 1) ? Wk : (w == 2) ? Wv : Wp;
  _Float16*  dst = (w == 0) ? WqT : (w == 1) ? WkT : (w == 2) ? WvT : WpT;
  dst[(long long)j * DD + i] = (_Float16)src[(long long)i * DD + j];
}

// ---------------- dist -> tanh(10*sqrt(max(.,0))) in place ----------------
__global__ __launch_bounds__(256) void dist_weight(
    float* __restrict__ s, const float* __restrict__ s2a, const float* __restrict__ s2b) {
  long long idx = (long long)blockIdx.x * 256 + threadIdx.x;
  long long b = idx / NSQ;
  long long rem = idx % NSQ;
  long long r = rem / NN, c = rem % NN;
  float dot = s[idx];
  float sqd = s2a[b * NN + r] + s2b[b * NN + c] - 2.0f * dot;
  s[idx] = tanhf(10.0f * sqrtf(fmaxf(sqd, 0.0f)));
}

// ---------------- per-row max of tanh-dist ----------------
__global__ __launch_bounds__(256) void row_max(const float* __restrict__ s, float* __restrict__ rmax) {
  __shared__ float sh[256];
  const long long row = blockIdx.x;          // 0..B*N-1
  const float* p = s + row * NN;
  const int t = threadIdx.x;
  float m = -3.4e38f;
#pragma unroll
  for (int i = 0; i < 8; ++i) m = fmaxf(m, p[t + i * 256]);
  float r = blockMax(m, sh);
  if (t == 0) rmax[row] = r;
}

// ---------------- softmax rows: write fp32 attn + f16 attn ----------------
__global__ __launch_bounds__(256) void softmax_row(
    const float* __restrict__ s, float* __restrict__ attnF, _Float16* __restrict__ attnH) {
  __shared__ float sh[256];
  const long long row = blockIdx.x;          // 0..B*N-1
  const float* p = s + row * NN;
  const int t = threadIdx.x;
  float v[8], m = -3.4e38f;
#pragma unroll
  for (int i = 0; i < 8; ++i) { v[i] = p[t + i * 256]; m = fmaxf(m, v[i]); }
  float mx = blockMax(m, sh);
  float lsum = 0.f;
#pragma unroll
  for (int i = 0; i < 8; ++i) { v[i] = expf(v[i] - mx); lsum += v[i]; }
  float tot = blockSum(lsum, sh);
  float inv = 1.0f / tot;
#pragma unroll
  for (int i = 0; i < 8; ++i) {
    float a = v[i] * inv;
    long long c = t + i * 256;
    attnF[row * NN + c] = a;
    attnH[row * NN + c] = (_Float16)a;
  }
}

// ---------------- host launch ----------------
extern "C" void kernel_launch(void* const* d_in, const int* in_sizes, int n_in,
                              void* d_out, int out_size, void* d_ws, size_t ws_size,
                              hipStream_t stream) {
  (void)in_sizes; (void)n_in; (void)out_size; (void)ws_size;
  const float* w2v = (const float*)d_in[0];
  const float* x   = (const float*)d_in[1];
  const float* gq  = (const float*)d_in[2];  const float* lbq = (const float*)d_in[3];
  const float* gk  = (const float*)d_in[4];  const float* lbk = (const float*)d_in[5];
  const float* gv  = (const float*)d_in[6];  const float* lbv = (const float*)d_in[7];
  const float* Wq  = (const float*)d_in[8];  const float* bq  = (const float*)d_in[9];
  const float* Wk  = (const float*)d_in[10]; const float* bk  = (const float*)d_in[11];
  const float* Wv  = (const float*)d_in[12]; const float* bv  = (const float*)d_in[13];
  const float* Wp  = (const float*)d_in[14]; const float* bp  = (const float*)d_in[15];
  const float* lamP = (const float*)d_in[16];

  float* outO = (float*)d_out;                    // [B,N,D]
  float* outA = outO + BN * DD;                   // [B,N,N]

  // -------- workspace carve --------
  char* ws = (char*)d_ws;
  size_t off = 0;
  auto carve = [&](size_t bytes) -> void* {
    void* p = ws + off;
    off = (off + bytes + 255) & ~(size_t)255;
    return p;
  };
  const size_t rowH = (size_t)BN * DD * sizeof(_Float16);
  _Float16* w2v_h = (_Float16*)carve(rowH);
  _Float16* x_h   = (_Float16*)carve(rowH);
  _Float16* lnq_h = (_Float16*)carve(rowH);
  _Float16* lnk_h = (_Float16*)carve(rowH);
  _Float16* lnv_h = (_Float16*)carve(rowH);
  _Float16* q_h   = (_Float16*)carve(rowH);
  _Float16* k_h   = (_Float16*)carve(rowH);
  _Float16* vT_h  = (_Float16*)carve(rowH);       // [B, D, N]
  _Float16* av_h  = (_Float16*)carve(rowH);       // [B, N, D]
  _Float16* attn_h = (_Float16*)carve((size_t)BB * NSQ * sizeof(_Float16));
  _Float16* WqT = (_Float16*)carve((size_t)DD * DD * sizeof(_Float16));
  _Float16* WkT = (_Float16*)carve((size_t)DD * DD * sizeof(_Float16));
  _Float16* WvT = (_Float16*)carve((size_t)DD * DD * sizeof(_Float16));
  _Float16* WpT = (_Float16*)carve((size_t)DD * DD * sizeof(_Float16));
  float* s2a   = (float*)carve((size_t)BN * sizeof(float));
  float* s2b   = (float*)carve((size_t)BN * sizeof(float));
  float* rmax  = (float*)carve((size_t)BN * sizeof(float));
  float* score = (float*)carve((size_t)BB * NSQ * sizeof(float));

  // -------- 1: prep weights (transpose -> f16) --------
  {
    long long tot = 4LL * DD * DD;
    prep_weights<<<dim3((unsigned)((tot + 255) / 256)), dim3(256), 0, stream>>>(
        Wq, Wk, Wv, Wp, WqT, WkT, WvT, WpT);
  }
  // -------- 2: LN + convert + sumsq --------
  ln_prep<<<dim3((unsigned)(2 * BN)), dim3(256), 0, stream>>>(
      w2v, x, gq, lbq, gk, lbk, gv, lbv, w2v_h, x_h, lnq_h, lnk_h, lnv_h, s2a, s2b);

  // -------- 3/4/5: q, k, v projections (flat M = B*N) --------
  {
    GemmP p{};
    p.K = DD; p.alpha = 1.0f;
    p.lda = DD; p.ldb = DD; p.sA = 0; p.sB = 0;
    dim3 g(DD / TN, (unsigned)(BN / TM), 1);
    // q
    p.A = lnq_h; p.Bt = WqT; p.bias = bq;
    p.outH = q_h; p.sH = 0; p.ldh = DD;
    gemm_wmma<EPI_BIAS_H><<<g, dim3(256), 0, stream>>>(p);
    // k
    p.A = lnk_h; p.Bt = WkT; p.bias = bk;
    p.outH = k_h;
    gemm_wmma<EPI_BIAS_H><<<g, dim3(256), 0, stream>>>(p);
    // v (stored transposed per batch: [B, D, N])
    GemmP pv{};
    pv.K = DD; pv.alpha = 1.0f; pv.lda = DD; pv.ldb = DD;
    pv.A = lnv_h; pv.Bt = WvT; pv.bias = bv;
    pv.outHT = vT_h; pv.sHT = (long long)DD * NN; pv.ldht = NN; pv.rowsPerBatchT = NN;
    gemm_wmma<EPI_BIAS_HT><<<g, dim3(256), 0, stream>>>(pv);
  }

  // -------- 6: raw dot GEMM  score = w2v . x^T  (batched) --------
  {
    GemmP p{};
    p.A = w2v_h; p.sA = (long long)NN * DD; p.lda = DD;
    p.Bt = x_h;  p.sB = (long long)NN * DD; p.ldb = DD;
    p.K = DD; p.alpha = 1.0f;
    p.outF = score; p.sF = NSQ; p.ldf = NN;
    dim3 g(NN / TN, NN / TM, BB);
    gemm_wmma<EPI_F32><<<g, dim3(256), 0, stream>>>(p);
  }
  // -------- 7: score -> tanh(10*dist) in place --------
  {
    long long tot = BB * NSQ;
    dist_weight<<<dim3((unsigned)(tot / 256)), dim3(256), 0, stream>>>(score, s2a, s2b);
  }
  // -------- 8: row max --------
  row_max<<<dim3((unsigned)BN), dim3(256), 0, stream>>>(score, rmax);

  // -------- 9: scores = (1-lam)*scale*(q.k^T) + lam*(w/rowmax + I) --------
  {
    GemmP p{};
    p.A = q_h; p.sA = (long long)NN * DD; p.lda = DD;
    p.Bt = k_h; p.sB = (long long)NN * DD; p.ldb = DD;
    p.K = DD;
    p.Cin = score; p.sC = NSQ; p.ldc = NN;
    p.rowmax = rmax; p.sRM = NN;
    p.lamP = lamP; p.scale = 1.0f / sqrtf((float)DD);
    p.outF = score; p.sF = NSQ; p.ldf = NN;   // in-place (exclusive tile ownership)
    dim3 g(NN / TN, NN / TM, BB);
    gemm_wmma<EPI_WMIX><<<g, dim3(256), 0, stream>>>(p);
  }
  // -------- 10: softmax rows -> attn (fp32 out) + attn_h (f16) --------
  softmax_row<<<dim3((unsigned)BN), dim3(256), 0, stream>>>(score, outA, attn_h);

  // -------- 11: av = attn . v   (batched, B operand = v^T) --------
  {
    GemmP p{};
    p.A = attn_h; p.sA = NSQ; p.lda = NN;
    p.Bt = vT_h;  p.sB = (long long)DD * NN; p.ldb = NN;
    p.K = NN; p.alpha = 1.0f;
    p.outH = av_h; p.sH = (long long)NN * DD; p.ldh = DD;
    dim3 g(DD / TN, NN / TM, BB);
    gemm_wmma<EPI_H><<<g, dim3(256), 0, stream>>>(p);
  }
  // -------- 12: out = av . Wp + bp  (flat M = B*N) --------
  {
    GemmP p{};
    p.A = av_h; p.lda = DD;
    p.Bt = WpT; p.ldb = DD;
    p.K = DD; p.alpha = 1.0f; p.bias = bp;
    p.outF = outO; p.sF = 0; p.ldf = DD;
    dim3 g(DD / TN, (unsigned)(BN / TM), 1);
    gemm_wmma<EPI_BIAS_F32><<<g, dim3(256), 0, stream>>>(p);
  }
}
